// CogViewEMAQuantizer_4724464026121
// MI455X (gfx1250) — matrix operand order repaired and checked by reading, not compiled, not run on hardware
//
#include <hip/hip_runtime.h>

#define DIM    256
#define NEMB   8192
#define NROWS  32768
#define DECAYF 0.99f
#define ONEMD  0.01f
#define EPSF   1e-5f
#define TILE_N 16
#define NTILES (NEMB / TILE_N)
#define LDS_STRIDE 264   // 256 + 8 bf16 pad -> lane col stride 132 dwords, conflict-free

typedef __attribute__((ext_vector_type(16))) __bf16 v16bf;
typedef __attribute__((ext_vector_type(8)))  float  v8f;

__device__ __forceinline__ __bf16 f2bf(float f) {
  union { float f; unsigned u; } c; c.f = f;
  unsigned u = c.u;
  unsigned r = (u + 0x7FFFu + ((u >> 16) & 1u)) >> 16;  // round to nearest even
  union { unsigned short s; __bf16 b; } o; o.s = (unsigned short)r;
  return o.b;
}

// Async memory->LDS copy of 32B (two b128 halves), tracked with ASYNCcnt.
// INST_OFFSET is added to BOTH the LDS and global addresses (ISA 08 §4.4),
// so one LDS-offset VGPR + one 64-bit address pair covers both halves.
__device__ __forceinline__ void async_copy32B(const __bf16* gsrc, __bf16* ldst) {
  unsigned l0 = (unsigned)(unsigned long long)(void*)ldst;  // addr[31:0] == LDS offset
  asm volatile("global_load_async_to_lds_b128 %0, %1, off"
               :: "v"(l0), "v"(gsrc) : "memory");
  asm volatile("global_load_async_to_lds_b128 %0, %1, off offset:16"
               :: "v"(l0), "v"(gsrc) : "memory");
}

// ---------------------------------------------------------------------------
// 1) Init: pre-scale EMA accumulators into the output buffers, zero scalars.
// ---------------------------------------------------------------------------
__global__ __launch_bounds__(256) void vq_init_kernel(
    const float* __restrict__ cluster_size, const float* __restrict__ embed_avg,
    float* __restrict__ cluster_new, float* __restrict__ embed_avg_new,
    float* __restrict__ scalars /* [0]=diffAcc [1]=n */) {
  int gid = blockIdx.x * 256 + threadIdx.x;
  if (gid < DIM * NEMB) embed_avg_new[gid] = DECAYF * embed_avg[gid];
  if (gid < NEMB)       cluster_new[gid]   = DECAYF * cluster_size[gid];
  if (gid == 0) { scalars[0] = 0.0f; scalars[1] = 0.0f; }
}

// ---------------------------------------------------------------------------
// 2) Prep: bf16 column-major codebook panel + per-code squared norms.
// ---------------------------------------------------------------------------
__global__ __launch_bounds__(256) void vq_prep_kernel(
    const float* __restrict__ embed, __bf16* __restrict__ embB,
    float* __restrict__ enorm) {
  int j = blockIdx.x * 256 + threadIdx.x;     // code index 0..8191
  float nrm = 0.0f;
  #pragma unroll 4
  for (int k = 0; k < DIM; ++k) {
    float v = embed[(size_t)k * NEMB + j];    // coalesced across j
    nrm = fmaf(v, v, nrm);
    embB[(size_t)j * DIM + k] = f2bf(v);
  }
  enorm[j] = nrm;
}

// ---------------------------------------------------------------------------
// 3) Argmin GEMM: score = enorm[j] - 2 * (x . e_j), running argmin per row.
//    8 waves/block, 32 rows/wave (2 M-subtiles share every B fragment).
//    B tiles double-buffered in LDS, filled by GLOBAL_LOAD_ASYNC_TO_LDS_B128.
// ---------------------------------------------------------------------------
__global__ __launch_bounds__(256) void vq_argmin_kernel(
    const float* __restrict__ x, const __bf16* __restrict__ embB,
    const float* __restrict__ enorm, int* __restrict__ embed_ind) {
  __shared__ __bf16 Bt[2][TILE_N * LDS_STRIDE];

  const int tid  = threadIdx.x;
  const int wave = tid >> 5;
  const int lane = tid & 31;
  const int m = lane & 15;     // row-within-subtile (A) / column slot (B,C)
  const int h = lane >> 4;     // lane half

  const int waveRow = blockIdx.x * 256 + wave * 32;

  // Preload A fragments: 16-bit A 16x32 k order used here:
  // lane half h holds K = 16h+{0..15} of row m (contiguous, matches B).
  v16bf A[2][8];
  #pragma unroll
  for (int s = 0; s < 2; ++s) {
    const float* rowp = x + (size_t)(waveRow + s * 16 + m) * DIM;
    #pragma unroll
    for (int kc = 0; kc < 8; ++kc) {
      v16bf a;
      #pragma unroll
      for (int half = 0; half < 2; ++half) {
        const float4* fp = (const float4*)(rowp + kc * 32 + 16 * h + 8 * half);
        float4 v0 = fp[0], v1 = fp[1];
        int b = 8 * half;
        a[b + 0] = f2bf(v0.x); a[b + 1] = f2bf(v0.y);
        a[b + 2] = f2bf(v0.z); a[b + 3] = f2bf(v0.w);
        a[b + 4] = f2bf(v1.x); a[b + 5] = f2bf(v1.y);
        a[b + 6] = f2bf(v1.z); a[b + 7] = f2bf(v1.w);
      }
      A[s][kc] = a;
    }
  }

  float bestV[2][8];
  int   bestI[2][8];
  #pragma unroll
  for (int s = 0; s < 2; ++s)
    #pragma unroll
    for (int r = 0; r < 8; ++r) { bestV[s][r] = 3.4e38f; bestI[s][r] = 0; }

  const int cCol = tid >> 4;   // column this thread stages (0..15)
  const int cSeg = tid & 15;   // 16-element K segment (32B)

  // Prologue: kick off tile 0 into buffer 0.
  async_copy32B(embB + ((size_t)(0 * TILE_N + cCol) * DIM + cSeg * 16),
                &Bt[0][cCol * LDS_STRIDE + cSeg * 16]);

  for (int nt = 0; nt < NTILES; ++nt) {
    const int cur = nt & 1;

    // Issue the NEXT tile's async copies into the other buffer (its readers
    // finished at the previous iteration's trailing barrier).
    if (nt + 1 < NTILES) {
      async_copy32B(embB + ((size_t)((nt + 1) * TILE_N + cCol) * DIM + cSeg * 16),
                    &Bt[1 - cur][cCol * LDS_STRIDE + cSeg * 16]);
      // 2 in flight for next tile; <=2 pending means current tile landed.
      asm volatile("s_wait_asynccnt 0x2" ::: "memory");
    } else {
      asm volatile("s_wait_asynccnt 0x0" ::: "memory");
    }
    __syncthreads();   // every wave's async copies for `cur` are visible

    v8f c0 = {}; v8f c1 = {};
    const __bf16* Bb = &Bt[cur][0];
    #pragma unroll
    for (int kc = 0; kc < 8; ++kc) {
      // B 32x16 bf16 layout: lanes 0-15 hold K=0..15, lanes 16-31 K=16..31
      // of column m -> contiguous 16 bf16 per lane in column-major LDS.
      union { uint4 u[2]; v16bf v; } bu;
      const uint4* bp = (const uint4*)(Bb + m * LDS_STRIDE + kc * 32 + h * 16);
      bu.u[0] = bp[0]; bu.u[1] = bp[1];
      v16bf b = bu.v;
      c0 = __builtin_amdgcn_wmma_f32_16x16x32_bf16(false, A[0][kc], false, b,
                                                   (short)0, c0, false, false);
      c1 = __builtin_amdgcn_wmma_f32_16x16x32_bf16(false, A[1][kc], false, b,
                                                   (short)0, c1, false, false);
    }

    const float en = enorm[nt * TILE_N + m];   // L2-resident, hidden under WMMA
    const int   j  = nt * TILE_N + m;
    #pragma unroll
    for (int r = 0; r < 8; ++r) {
      float s0 = fmaf(-2.0f, c0[r], en);
      if (s0 < bestV[0][r]) { bestV[0][r] = s0; bestI[0][r] = j; }
      float s1 = fmaf(-2.0f, c1[r], en);
      if (s1 < bestV[1][r]) { bestV[1][r] = s1; bestI[1][r] = j; }
    }
    __syncthreads();   // readers done before this buffer is re-filled
  }

  // Cross-lane argmin within each 16-lane half (ties -> lowest index).
  #pragma unroll
  for (int s = 0; s < 2; ++s) {
    #pragma unroll
    for (int r = 0; r < 8; ++r) {
      float v = bestV[s][r];
      int   i = bestI[s][r];
      #pragma unroll
      for (int off = 8; off > 0; off >>= 1) {
        float ov = __shfl_xor(v, off, 16);
        int   oi = __shfl_xor(i, off, 16);
        if (ov < v || (ov == v && oi < i)) { v = ov; i = oi; }
      }
      if (m == 0) embed_ind[waveRow + s * 16 + h * 8 + r] = i;  // C row = r + 8h
    }
  }
}

// ---------------------------------------------------------------------------
// 4) Gather codewords, commitment loss partials, EMA segment sums (atomics).
// ---------------------------------------------------------------------------
__global__ __launch_bounds__(DIM) void vq_gather_kernel(
    const float* __restrict__ x, const float* __restrict__ embed,
    const int* __restrict__ ind, float* __restrict__ qout,
    float* __restrict__ cluster_new, float* __restrict__ embed_avg_new,
    float* __restrict__ scalars) {
  __shared__ float red[DIM];
  const int row = blockIdx.x;
  const int k   = threadIdx.x;
  const int j   = ind[row];
  float qv = embed[(size_t)k * NEMB + j];
  float xv = x[(size_t)row * DIM + k];
  qout[(size_t)row * DIM + k] = qv;            // quantize_st == quantize fwd
  float d = qv - xv;
  red[k] = d * d;
  atomicAdd(&embed_avg_new[(size_t)k * NEMB + j], ONEMD * xv);
  __syncthreads();
  #pragma unroll
  for (int s = DIM / 2; s > 0; s >>= 1) {
    if (k < s) red[k] += red[k + s];
    __syncthreads();
  }
  if (k == 0) {
    atomicAdd(&scalars[0], red[0]);
    atomicAdd(&cluster_new[j], ONEMD);
  }
}

// ---------------------------------------------------------------------------
// 5) n = sum(cluster_size_new); diff = diffAcc / (N*DIM)
// ---------------------------------------------------------------------------
__global__ __launch_bounds__(256) void vq_finalize_n_kernel(
    const float* __restrict__ cluster_new, float* __restrict__ scalars,
    float* __restrict__ diff_out) {
  __shared__ float red[256];
  const int t = threadIdx.x;
  float s = 0.0f;
  for (int j = t; j < NEMB; j += 256) s += cluster_new[j];
  red[t] = s;
  __syncthreads();
  #pragma unroll
  for (int k = 128; k > 0; k >>= 1) {
    if (t < k) red[t] += red[t + k];
    __syncthreads();
  }
  if (t == 0) {
    scalars[1]  = red[0];
    diff_out[0] = scalars[0] * (1.0f / ((float)NROWS * (float)DIM));
  }
}

// ---------------------------------------------------------------------------
// 6) embed_new = embed_avg_new / Laplace-smoothed cluster size
// ---------------------------------------------------------------------------
__global__ __launch_bounds__(256) void vq_finalize_embed_kernel(
    const float* __restrict__ embed_avg_new, const float* __restrict__ cluster_new,
    const float* __restrict__ scalars, float* __restrict__ embed_new) {
  int gid = blockIdx.x * 256 + threadIdx.x;
  if (gid >= DIM * NEMB) return;
  int j = gid & (NEMB - 1);
  float n   = scalars[1];
  float csn = cluster_new[j];
  float cs  = (csn + EPSF) / (n + (float)NEMB * EPSF) * n;
  embed_new[gid] = embed_avg_new[gid] / cs;
}

// ---------------------------------------------------------------------------
extern "C" void kernel_launch(void* const* d_in, const int* in_sizes, int n_in,
                              void* d_out, int out_size, void* d_ws, size_t ws_size,
                              hipStream_t stream) {
  const float* x         = (const float*)d_in[0];   // [8,64,64,256]
  const float* embed     = (const float*)d_in[1];   // [256,8192]
  const float* cluster   = (const float*)d_in[2];   // [8192]
  const float* embed_avg = (const float*)d_in[3];   // [256,8192]

  // Output tuple, concatenated flat (float units):
  float* out          = (float*)d_out;
  float* q_out        = out;                        // 2097152
  float* diff_out     = out + 2097152;              // 1
  int*   ind_out      = (int*)(out + 2097153);      // 32768 (int32)
  float* embed_new    = out + 2129921;              // 2097152
  float* cluster_new  = out + 4227073;              // 8192
  float* embedavg_new = out + 4235265;              // 2097152

  // Workspace: bf16 codebook panel (4MB) + norms (32KB) + 2 scalars
  char*   w       = (char*)d_ws;
  __bf16* embB    = (__bf16*)w;
  float*  enorm   = (float*)(w + (size_t)NEMB * DIM * 2);
  float*  scalars = (float*)(w + (size_t)NEMB * DIM * 2 + (size_t)NEMB * 4);

  vq_init_kernel<<<(DIM * NEMB + 255) / 256, 256, 0, stream>>>(
      cluster, embed_avg, cluster_new, embedavg_new, scalars);
  vq_prep_kernel<<<NEMB / 256, 256, 0, stream>>>(embed, embB, enorm);
  vq_argmin_kernel<<<NROWS / 256, 256, 0, stream>>>(x, embB, enorm, ind_out);
  vq_gather_kernel<<<NROWS, DIM, 0, stream>>>(
      x, embed, ind_out, q_out, cluster_new, embedavg_new, scalars);
  vq_finalize_n_kernel<<<1, 256, 0, stream>>>(cluster_new, scalars, diff_out);
  vq_finalize_embed_kernel<<<(DIM * NEMB) / 256, 256, 0, stream>>>(
      embedavg_new, cluster_new, scalars, embed_new);
}
